// model2_42992622633180
// MI455X (gfx1250) — matrix operand order repaired and checked by reading, not compiled, not run on hardware
//
#include <hip/hip_runtime.h>
#include <hip/hip_bf16.h>

typedef __attribute__((ext_vector_type(16))) _Float16 v16h;
typedef __attribute__((ext_vector_type(8)))  float    v8f;

#define H_      51
#define COND_   1024
#define TOTAL_  1088
#define BATCH_  8192
#define MROWS   32           // batch rows per block
#define NP      256          // padded gate width: 4 gates x 64
#define KSTEPS  4            // K = 128 = [input|64, hidden|64], 4 x 32
#define THREADS 256          // 8 waves = 4 gates x 2 M-tiles

// LDS region element counts
#define WFRAG_HALVES (3*KSTEPS*16*32*16)  // 98304 halves = 192 KB (weights, frag layout)
#define AFRAG_HALVES (3*2*KSTEPS*32*16)   // 12288 halves (activations, frag layout)
#define CBUF_F       (3*MROWS*64)         // 6144 floats (cell state)
#define GBUF_F       (MROWS*NP)           // 8192 floats (gate scratch)
#define H2_F         (MROWS*64)           // 2048 floats (top-layer h, f32)
#define BIAS_F       (3*NP)               // 768 floats
#define WLIN_F       64

#define SMEM_BYTES (WFRAG_HALVES*2 + AFRAG_HALVES*2 + \
                    (CBUF_F + GBUF_F + H2_F + BIAS_F + WLIN_F)*4 + 16)

__device__ __forceinline__ float sigf(float x) {
  return 1.0f / (1.0f + __expf(-x));
}
__device__ __forceinline__ float tanh_f(float x) {
  float e = __expf(-2.0f * fabsf(x));
  float t = (1.0f - e) / (1.0f + e);
  return copysignf(t, x);
}

extern "C" __global__ __launch_bounds__(THREADS)
void lstm_stack_kernel(const float* __restrict__ input,
                       const float* __restrict__ time_idx,
                       const float* __restrict__ Wih0, const float* __restrict__ Whh0,
                       const float* __restrict__ bih0, const float* __restrict__ bhh0,
                       const float* __restrict__ Wih1, const float* __restrict__ Whh1,
                       const float* __restrict__ bih1, const float* __restrict__ bhh1,
                       const float* __restrict__ Wih2, const float* __restrict__ Whh2,
                       const float* __restrict__ bih2, const float* __restrict__ bhh2,
                       const float* __restrict__ Wlin, const float* __restrict__ blin,
                       float* __restrict__ out)
{
  extern __shared__ __align__(16) char smem_raw[];
  _Float16* wfrag = (_Float16*)smem_raw;
  _Float16* afrag = wfrag + WFRAG_HALVES;
  float* cbuf   = (float*)(afrag + AFRAG_HALVES);
  float* gbuf   = cbuf + CBUF_F;
  float* h2buf  = gbuf + GBUF_F;
  float* biasc  = h2buf + H2_F;
  float* wlin   = biasc + BIAS_F;
  float* blin_s = wlin + WLIN_F;

  const int tid   = threadIdx.x;
  const int bbase = blockIdx.x * MROWS;
  const int wv    = tid >> 5;
  const int lane  = tid & 31;
  const int gw    = wv & 3;   // gate owned by this wave (i,f,g,o)
  const int mt    = wv >> 2;  // M-tile (0..1)

  // ---------- one-time setup: weights -> f16 WMMA B-fragments in LDS ----------
  // wfrag element e = (((l*4+ks)*16+nt)*32+lane)*16 + i
  // B-fragment layout: lane holds column n = nt*16 + (lane&15),
  //                    k' = (lane>>4)*16 + i  (i = 0..15), k = ks*32 + k'
  for (int e = tid; e < WFRAG_HALVES; e += THREADS) {
    int i  = e & 15;
    int ln = (e >> 4) & 31;
    int nt = (e >> 9) & 15;
    int ks = (e >> 13) & 3;
    int l  = e >> 15;
    int n  = nt * 16 + (ln & 15);
    int k  = ks * 32 + ((ln >> 4) * 16 + i);
    int g = n >> 6, j = n & 63;
    float w = 0.0f;
    if (j < H_) {
      int r = g * H_ + j;   // PyTorch gate order i,f,g,o
      if (k < 64) {         // input-part columns
        if (l == 0) { if (k < 2)  w = Wih0[r*2  + k]; }
        else {
          const float* Wih = (l == 1) ? Wih1 : Wih2;
          if (k < H_) w = Wih[r*H_ + k];
        }
      } else {              // hidden-part columns
        int kk = k - 64;
        const float* Whh = (l == 0) ? Whh0 : (l == 1) ? Whh1 : Whh2;
        if (kk < H_) w = Whh[r*H_ + kk];
      }
    }
    wfrag[e] = (_Float16)w;
  }
  // combined bias per padded gate column
  for (int e = tid; e < BIAS_F; e += THREADS) {
    int l = e >> 8, n = e & 255;
    int g = n >> 6, j = n & 63;
    float v = 0.0f;
    if (j < H_) {
      int r = g * H_ + j;
      const float* bi = (l == 0) ? bih0 : (l == 1) ? bih1 : bih2;
      const float* bh = (l == 0) ? bhh0 : (l == 1) ? bhh1 : bhh2;
      v = bi[r] + bh[r];
    }
    biasc[e] = v;
  }
  // zero activation fragments (h0 = 0; padding lanes stay 0 forever) and cell state
  for (int e = tid; e < AFRAG_HALVES; e += THREADS) afrag[e] = (_Float16)0.0f;
  for (int e = tid; e < CBUF_F; e += THREADS) cbuf[e] = 0.0f;
  if (tid < WLIN_F) wlin[tid] = (tid < H_) ? Wlin[tid] : 0.0f;
  if (tid == 0) blin_s[0] = blin[0];

  float prev_out = 0.0f;  // autoregressive feedback (threads 0..31 only)
  __syncthreads();

  // ---------- sequential scan over 1088 time steps ----------
  for (int t = 0; t < TOTAL_; ++t) {
    // stage x_t = [value, time_feat] into layer-0 input fragment (k'=0,1 of kstep 0)
    if (tid < MROWS) {
      int m = tid;
      float x0 = (t < COND_) ? input[(bbase + m)*COND_ + t] : prev_out;
      float x1 = time_idx[(bbase + m)*TOTAL_ + t];
      _Float16* a0 = afrag + (((m >> 4) * KSTEPS) * 32 + (m & 15)) * 16;
      a0[0] = (_Float16)x0;
      a0[1] = (_Float16)x1;
    }
    __syncthreads();

    for (int l = 0; l < 3; ++l) {
      // ---- WMMA: gates(32 x 256) = A(32 x 128) * W(128 x 256) + bias ----
      const _Float16* ab = afrag + ((l*2 + mt) * KSTEPS) * 512 + lane * 16;
      v16h a0 = *(const v16h*)(ab);
      v16h a1 = *(const v16h*)(ab + 512);
      v16h a2 = *(const v16h*)(ab + 1024);
      v16h a3 = *(const v16h*)(ab + 1536);
      #pragma unroll
      for (int nt4 = 0; nt4 < 4; ++nt4) {
        int nt = gw * 4 + nt4;
        float bv = biasc[l*NP + nt*16 + (lane & 15)];
        v8f acc = {bv, bv, bv, bv, bv, bv, bv, bv};
        const _Float16* bb = wfrag + ((l*64 + nt) * 32 + lane) * 16;
        v16h b0 = *(const v16h*)(bb);
        acc = __builtin_amdgcn_wmma_f32_16x16x32_f16(false, a0, false, b0, (short)0, acc, false, false);
        v16h b1 = *(const v16h*)(bb + 16*512);
        acc = __builtin_amdgcn_wmma_f32_16x16x32_f16(false, a1, false, b1, (short)0, acc, false, false);
        v16h b2 = *(const v16h*)(bb + 32*512);
        acc = __builtin_amdgcn_wmma_f32_16x16x32_f16(false, a2, false, b2, (short)0, acc, false, false);
        v16h b3 = *(const v16h*)(bb + 48*512);
        acc = __builtin_amdgcn_wmma_f32_16x16x32_f16(false, a3, false, b3, (short)0, acc, false, false);
        // C/D layout: lane holds column n, rows r+8*(lane>>4)
        int n  = nt*16 + (lane & 15);
        int mb = mt*16 + 8*(lane >> 4);
        #pragma unroll
        for (int r = 0; r < 8; ++r)
          gbuf[(mb + r)*NP + n] = acc[r];
      }
      __syncthreads();

      // ---- elementwise LSTM cell; write new h back in fragment layout ----
      #pragma unroll
      for (int it = 0; it < (MROWS*64)/THREADS; ++it) {
        int e = tid + it*THREADS;
        int j = e & 63, m = e >> 6;
        if (j < H_) {
          float gi = gbuf[m*NP +       j];
          float gf = gbuf[m*NP +  64 + j];
          float gg = gbuf[m*NP + 128 + j];
          float go = gbuf[m*NP + 192 + j];
          float* cp = cbuf + (l*MROWS + m)*64 + j;
          float c  = *cp;
          float cn = sigf(gf)*c + sigf(gi)*tanh_f(gg);
          *cp = cn;
          float hn = sigf(go)*tanh_f(cn);
          _Float16 h16 = (_Float16)hn;
          // A-fragment layout: lane = 16*hh + (m&15); hh=k'>>4; i=k'&15
          int kp  = j & 31;
          int hh  = kp >> 4;
          int ii  = kp & 15;
          int ln2 = (hh << 4) + (m & 15);
          int mtm = m >> 4;
          // own hidden part (ksteps 2..3) for next time step
          afrag[(((l*2 + mtm)*KSTEPS + 2 + (j >> 5))*32 + ln2)*16 + ii] = h16;
          if (l < 2)  // input part (ksteps 0..1) of next layer
            afrag[((((l+1)*2 + mtm)*KSTEPS + (j >> 5))*32 + ln2)*16 + ii] = h16;
          else
            h2buf[m*64 + j] = hn;
        }
      }
      __syncthreads();
    }

    // ---- linear head: out = h2 @ Wlin^T + blin ----
    if (tid < MROWS) {
      int m = tid;
      float s = blin_s[0];
      #pragma unroll
      for (int j = 0; j < H_; ++j) s += h2buf[m*64 + j] * wlin[j];
      out[(bbase + m)*TOTAL_ + t] = s;
      prev_out = s;
    }
    __syncthreads();
  }
}

extern "C" void kernel_launch(void* const* d_in, const int* in_sizes, int n_in,
                              void* d_out, int out_size, void* d_ws, size_t ws_size,
                              hipStream_t stream) {
  (void)in_sizes; (void)n_in; (void)out_size; (void)d_ws; (void)ws_size;
  const float* input    = (const float*)d_in[0];
  const float* time_idx = (const float*)d_in[1];
  const float* Wih0 = (const float*)d_in[2];
  const float* Whh0 = (const float*)d_in[3];
  const float* bih0 = (const float*)d_in[4];
  const float* bhh0 = (const float*)d_in[5];
  const float* Wih1 = (const float*)d_in[6];
  const float* Whh1 = (const float*)d_in[7];
  const float* bih1 = (const float*)d_in[8];
  const float* bhh1 = (const float*)d_in[9];
  const float* Wih2 = (const float*)d_in[10];
  const float* Whh2 = (const float*)d_in[11];
  const float* bih2 = (const float*)d_in[12];
  const float* bhh2 = (const float*)d_in[13];
  const float* Wlin = (const float*)d_in[14];
  const float* blin = (const float*)d_in[15];
  float* out = (float*)d_out;

  // 290 KB dynamic LDS: legal on CDNA5 (320 KB per workgroup), one block/WGP
  (void)hipFuncSetAttribute((const void*)lstm_stack_kernel,
                            hipFuncAttributeMaxDynamicSharedMemorySize, SMEM_BYTES);
  dim3 grid(BATCH_ / MROWS);   // 256 independent blocks, each owns 32 batch rows
  dim3 block(THREADS);
  lstm_stack_kernel<<<grid, block, SMEM_BYTES, stream>>>(
      input, time_idx, Wih0, Whh0, bih0, bhh0, Wih1, Whh1, bih1, bhh1,
      Wih2, Whh2, bih2, bhh2, Wlin, blin, out);
}
